// WavenetDecoder_75926431859034
// MI455X (gfx1250) — compile-verified, matrix operand order checked by reading, not run on hardware
//
#include <hip/hip_runtime.h>
#include <hip/hip_bf16.h>

// ---------------------------------------------------------------------------
// WaveNet decoder on MI455X (gfx1250), bf16 WMMA (v_wmma_f32_16x16x32_bf16).
// Flat-time formulation: dilate-reshape trick == dilated conv with zero pad d.
// Round 4: A-tiles fed by GLOBAL_LOAD_ASYNC_TO_LDS_B128 (ASYNCcnt path),
// double-buffered LDS pipeline, tap-split branch-free fills.
// ---------------------------------------------------------------------------

typedef __attribute__((ext_vector_type(16))) __bf16 v16bf;
typedef __attribute__((ext_vector_type(8)))  float  v8f;

#define L_LEN   4096
#define RW_C    256
#define SW_C    128
#define CDIM    16
#define NL_C    30
#define KTOT_A  800   // 3*256 taps + 16 cond + 16 zero pad (25 chunks of 32)
#define LDA     40    // LDS row stride in bf16 (80 B, 16B-aligned, 20 banks)

union V16U { v16bf v; uint4 q[2]; };
union Q8U  { uint4 q; __bf16 h[8]; };

// Fragment load: elements e<8 -> K = 8h+e ; e>=8 -> K = 8h+16+(e-8)
// => two contiguous 16-byte runs at byte offsets 16h and 16h+32 of the row.
__device__ __forceinline__ v16bf frag_ld(const __bf16* rowbase, int h) {
    const uint4* p = (const uint4*)rowbase;
    V16U u;
    u.q[0] = p[h];
    u.q[1] = p[h + 2];
    return u.v;
}

// LDS byte offset of a __shared__ address (its low 32 bits)
__device__ __forceinline__ unsigned lds_off(const void* p) {
    return (unsigned)(unsigned long long)p;
}

// Async global->LDS 16B per lane (tracked by ASYNCcnt; no VGPR staging).
__device__ __forceinline__ void async_ld_b128(unsigned lds, const void* g) {
    asm volatile("global_load_async_to_lds_b128 %0, %1, off"
                 :: "v"(lds), "v"(g) : "memory");
}
__device__ __forceinline__ void wait_async0() {
#if __has_builtin(__builtin_amdgcn_s_wait_asynccnt)
    __builtin_amdgcn_s_wait_asynccnt(0);
#else
    asm volatile("s_wait_asynccnt 0x0" ::: "memory");
#endif
}

// ---------------------------------------------------------------------------
// Weight packing (once per launch): fp32 -> bf16, WMMA K-major layouts.
// ---------------------------------------------------------------------------
__global__ __launch_bounds__(256) void wn_pack_gau(
    const float* __restrict__ filter_w, const float* __restrict__ gate_w,
    const float* __restrict__ fcond_w,  const float* __restrict__ gcond_w,
    __bf16* __restrict__ Wf, __bf16* __restrict__ Wg)
{
    int idx = blockIdx.x * 256 + threadIdx.x;      // 30*256*800 = 6,144,000
    int kk = idx % KTOT_A;
    int t  = idx / KTOT_A;
    int m  = t % RW_C;
    int i  = t / RW_C;
    float vf = 0.f, vg = 0.f;
    if (kk < 768) {
        int tap = kk >> 8, c2 = kk & 255;          // K = tap*256 + c'
        size_t o = (((size_t)i * RW_C + m) * RW_C + c2) * 3 + tap;
        vf = filter_w[o];
        vg = gate_w[o];
    } else if (kk < 784) {
        size_t o = ((size_t)i * RW_C + m) * CDIM + (kk - 768);
        vf = fcond_w[o];
        vg = gcond_w[o];
    }
    Wf[idx] = (__bf16)vf;
    Wg[idx] = (__bf16)vg;
}

__global__ __launch_bounds__(256) void wn_pack_post(
    const float* __restrict__ feat_w, const float* __restrict__ skip_w,
    __bf16* __restrict__ Wp)
{
    int idx = blockIdx.x * 256 + threadIdx.x;      // 30*384*256 = 2,949,120
    int kk = idx & 255;
    int t  = idx >> 8;
    int m  = t % (RW_C + SW_C);
    int i  = t / (RW_C + SW_C);
    float v = (m < RW_C) ? feat_w[((size_t)i * RW_C + m) * RW_C + kk]
                         : skip_w[((size_t)i * SW_C + (m - RW_C)) * RW_C + kk];
    Wp[idx] = (__bf16)v;
}

__global__ __launch_bounds__(256) void wn_cvt(
    const float* __restrict__ src, __bf16* __restrict__ dst, int n)
{
    int idx = blockIdx.x * 256 + threadIdx.x;
    if (idx < n) dst[idx] = (__bf16)src[idx];
}

// ---------------------------------------------------------------------------
// init: feat[c,u] = sum_k init_w[c,k] * x[u+k-1] + init_b[c]  (pad 1)
// ---------------------------------------------------------------------------
__global__ __launch_bounds__(256) void wn_init(
    const float* __restrict__ x, const float* __restrict__ init_w,
    const float* __restrict__ init_b,
    float* __restrict__ feat_f32, __bf16* __restrict__ feat_bf)
{
    int idx = blockIdx.x * 256 + threadIdx.x;      // over 256*L
    int c = idx >> 12;
    int u = idx & (L_LEN - 1);
    float acc = init_b[c];
    #pragma unroll
    for (int k = 0; k < 3; ++k) {
        int uu = u + k - 1;
        if (uu >= 0 && uu < L_LEN) acc += init_w[c * 3 + k] * x[uu];
    }
    feat_f32[idx] = acc;
    feat_bf[idx]  = (__bf16)acc;
}

// ---------------------------------------------------------------------------
// Layer kernel 1: gated activation unit. Tile 64 rows x 64 time, 8 waves.
// Chunk 0 = conditioning, chunks 1..24 = taps (p: tap = p>>3, c2b=(p&7)*32).
// A tiles: async global->LDS.  B tile: regs + transpose scatter.
// ---------------------------------------------------------------------------
template<bool FAST>
__device__ __forceinline__ void gau_body(
    __bf16 (*Af)[64][LDA], __bf16 (*Ag)[64][LDA], __bf16 (*Bt)[64][LDA],
    const __bf16* __restrict__ wf, const __bf16* __restrict__ wg,
    const __bf16* __restrict__ feat_bf, const __bf16* __restrict__ cond_bf,
    __bf16* __restrict__ r_bf, int dil, int u0, int m0)
{
    const int tid  = threadIdx.x;
    const int lane = tid & 31;
    const int wave = tid >> 5;
    const int row16   = (wave & 3) * 16;
    const int colHalf = (wave >> 2) * 32;

    const int a_ml  = tid >> 2;            // 64 rows
    const int a_kl0 = (tid & 3) * 8;       // 4 x 8 K
    const int b_kl  = tid >> 3;            // 32 K rows
    const int b_n0  = (tid & 7) * 8;       // 8 x 8 n

    const __bf16* wfp = wf + (size_t)(m0 + a_ml) * KTOT_A + a_kl0;
    const __bf16* wgp = wg + (size_t)(m0 + a_ml) * KTOT_A + a_kl0;

    const unsigned ldsAf0 = lds_off(&Af[0][a_ml][a_kl0]);
    const unsigned ldsAf1 = lds_off(&Af[1][a_ml][a_kl0]);
    const unsigned ldsAg0 = lds_off(&Ag[0][a_ml][a_kl0]);
    const unsigned ldsAg1 = lds_off(&Ag[1][a_ml][a_kl0]);

    v8f accF0 = {}, accF1 = {}, accG0 = {}, accG1 = {};

    auto asyncA = [&](int buf, int kk0) {
        async_ld_b128(buf ? ldsAf1 : ldsAf0, wfp + kk0);
        async_ld_b128(buf ? ldsAg1 : ldsAg0, wgp + kk0);
    };
    auto store_B = [&](int buf, const Q8U& tb) {
        #pragma unroll
        for (int j = 0; j < 8; ++j) Bt[buf][b_n0 + j][b_kl] = tb.h[j];
    };
    auto fetch_B = [&](int tap, int c2b, Q8U& tb) {
        if (FAST) {
            tb.q = *(const uint4*)(feat_bf + (size_t)(c2b + b_kl) * L_LEN
                                   + (u0 + b_n0 + (tap - 1) * dil));
        } else {
            const __bf16* src = feat_bf + (size_t)(c2b + b_kl) * L_LEN;
            int uu0 = u0 + b_n0 + (tap - 1) * dil;
            #pragma unroll
            for (int j = 0; j < 8; ++j) {
                int uu = uu0 + j;
                tb.h[j] = ((unsigned)uu < (unsigned)L_LEN) ? src[uu] : (__bf16)0.f;
            }
        }
    };
    auto compute = [&](int buf) {
        const int hh = lane >> 4;
        v16bf af = frag_ld(&Af[buf][row16 + (lane & 15)][0], hh);
        v16bf ag = frag_ld(&Ag[buf][row16 + (lane & 15)][0], hh);
        v16bf b0 = frag_ld(&Bt[buf][colHalf + (lane & 15)][0], hh);
        v16bf b1 = frag_ld(&Bt[buf][colHalf + 16 + (lane & 15)][0], hh);
        accF0 = __builtin_amdgcn_wmma_f32_16x16x32_bf16(false, af, false, b0, (short)0, accF0, false, false);
        accF1 = __builtin_amdgcn_wmma_f32_16x16x32_bf16(false, af, false, b1, (short)0, accF1, false, false);
        accG0 = __builtin_amdgcn_wmma_f32_16x16x32_bf16(false, ag, false, b0, (short)0, accG0, false, false);
        accG1 = __builtin_amdgcn_wmma_f32_16x16x32_bf16(false, ag, false, b1, (short)0, accG1, false, false);
    };

    // ---- prologue: conditioning chunk (K = 768..799) into buffer 0 ----
    {
        asyncA(0, 768);
        Q8U tb;
        if (b_kl < CDIM) tb.q = *(const uint4*)(cond_bf + (size_t)b_kl * L_LEN + u0 + b_n0);
        else             tb.q = uint4{0u, 0u, 0u, 0u};
        store_B(0, tb);
        wait_async0();
    }
    __syncthreads();

    // ---- pipeline over 24 tap chunks ----
    for (int p = 0; p < 24; ++p) {
        const int tap = p >> 3;
        const int c2b = (p & 7) * 32;
        asyncA((p + 1) & 1, tap * 256 + c2b);  // A for chunk p+1 -> other buffer
        Q8U tb;
        fetch_B(tap, c2b, tb);                 // B regs for chunk p+1
        compute(p & 1);                        // WMMA on chunk p
        store_B((p + 1) & 1, tb);
        wait_async0();
        __syncthreads();
    }
    compute(0);                                // chunk 24 (stored at p=23)

    // ---- epilogue: r = sigmoid(f) * tanh(g) ----
    const int hh = lane >> 4;
    const int nc = lane & 15;
    #pragma unroll
    for (int r = 0; r < 8; ++r) {
        int m = m0 + row16 + 8 * hh + r;
        int u = u0 + colHalf + nc;
        float rr0 = (1.f / (1.f + __expf(-accF0[r]))) * tanhf(accG0[r]);
        r_bf[(size_t)m * L_LEN + u] = (__bf16)rr0;
        float rr1 = (1.f / (1.f + __expf(-accF1[r]))) * tanhf(accG1[r]);
        r_bf[(size_t)m * L_LEN + u + 16] = (__bf16)rr1;
    }
}

__global__ __launch_bounds__(256) void wn_gau(
    const __bf16* __restrict__ Wf, const __bf16* __restrict__ Wg,
    const __bf16* __restrict__ feat_bf, const __bf16* __restrict__ cond_bf,
    __bf16* __restrict__ r_bf, int layer, int dil)
{
    __shared__ __attribute__((aligned(16))) __bf16 Af[2][64][LDA];
    __shared__ __attribute__((aligned(16))) __bf16 Ag[2][64][LDA];
    __shared__ __attribute__((aligned(16))) __bf16 Bt[2][64][LDA];

    const int u0 = blockIdx.x * 64;
    const int m0 = blockIdx.y * 64;
    const __bf16* wf = Wf + (size_t)layer * RW_C * KTOT_A;
    const __bf16* wg = Wg + (size_t)layer * RW_C * KTOT_A;

    const bool fast = ((dil & 7) == 0) && (u0 >= dil) && (u0 + 64 + dil <= L_LEN);
    if (fast) gau_body<true >(Af, Ag, Bt, wf, wg, feat_bf, cond_bf, r_bf, dil, u0, m0);
    else      gau_body<false>(Af, Ag, Bt, wf, wg, feat_bf, cond_bf, r_bf, dil, u0, m0);
}

// ---------------------------------------------------------------------------
// Layer kernel 2: [feat_w;skip_w] @ r with fused residual add + skip accum.
// ---------------------------------------------------------------------------
__global__ __launch_bounds__(256) void wn_post(
    const __bf16* __restrict__ Wp, const __bf16* __restrict__ r_bf,
    float* __restrict__ feat_f32, __bf16* __restrict__ feat_bf,
    float* __restrict__ skip_f32, int layer)
{
    __shared__ __attribute__((aligned(16))) __bf16 As[2][64][LDA];
    __shared__ __attribute__((aligned(16))) __bf16 Bt[2][64][LDA];

    const int tid  = threadIdx.x;
    const int lane = tid & 31;
    const int wave = tid >> 5;
    const int row16   = (wave & 3) * 16;
    const int colHalf = (wave >> 2) * 32;
    const int u0 = blockIdx.x * 64;
    const int m0 = blockIdx.y * 64;

    const int a_ml  = tid >> 2;
    const int a_kl0 = (tid & 3) * 8;
    const int b_kl  = tid >> 3;
    const int b_n0  = (tid & 7) * 8;

    const __bf16* wpp = Wp + (size_t)layer * (RW_C + SW_C) * RW_C
                        + (size_t)(m0 + a_ml) * RW_C + a_kl0;
    const __bf16* rbp = r_bf + (size_t)b_kl * L_LEN + u0 + b_n0;

    const unsigned ldsA0 = lds_off(&As[0][a_ml][a_kl0]);
    const unsigned ldsA1 = lds_off(&As[1][a_ml][a_kl0]);

    v8f acc0 = {}, acc1 = {};

    auto store_B = [&](int buf, const Q8U& tb) {
        #pragma unroll
        for (int j = 0; j < 8; ++j) Bt[buf][b_n0 + j][b_kl] = tb.h[j];
    };
    auto compute = [&](int buf) {
        const int hh = lane >> 4;
        v16bf a  = frag_ld(&As[buf][row16 + (lane & 15)][0], hh);
        v16bf b0 = frag_ld(&Bt[buf][colHalf + (lane & 15)][0], hh);
        v16bf b1 = frag_ld(&Bt[buf][colHalf + 16 + (lane & 15)][0], hh);
        acc0 = __builtin_amdgcn_wmma_f32_16x16x32_bf16(false, a, false, b0, (short)0, acc0, false, false);
        acc1 = __builtin_amdgcn_wmma_f32_16x16x32_bf16(false, a, false, b1, (short)0, acc1, false, false);
    };

    {   // prologue: chunk 0
        async_ld_b128(ldsA0, wpp);
        Q8U tb; tb.q = *(const uint4*)(rbp);
        store_B(0, tb);
        wait_async0();
    }
    __syncthreads();
    for (int p = 0; p < 7; ++p) {
        async_ld_b128((p + 1) & 1 ? ldsA1 : ldsA0, wpp + (p + 1) * 32);
        Q8U tb;  tb.q = *(const uint4*)(rbp + (size_t)(p + 1) * 32 * L_LEN);
        compute(p & 1);
        store_B((p + 1) & 1, tb);
        wait_async0();
        __syncthreads();
    }
    compute(1);                            // chunk 7

    const int hh = lane >> 4;
    const int nc = lane & 15;
    #pragma unroll
    for (int r = 0; r < 8; ++r) {
        int m = m0 + row16 + 8 * hh + r;
        int u = u0 + colHalf + nc;
        if (m < RW_C) {
            size_t o = (size_t)m * L_LEN + u;
            float v0 = acc0[r] + feat_f32[o];
            feat_f32[o] = v0; feat_bf[o] = (__bf16)v0;
            float v1 = acc1[r] + feat_f32[o + 16];
            feat_f32[o + 16] = v1; feat_bf[o + 16] = (__bf16)v1;
        } else {
            size_t o = (size_t)(m - RW_C) * L_LEN + u;
            skip_f32[o]      += acc0[r];
            skip_f32[o + 16] += acc1[r];
        }
    }
}

// ---------------------------------------------------------------------------
// Small GEMM: out[m,u] = Apk(MxK,bf16) @ B(KxL) + bias[m].
// B source: bf16 buffer, or fp32 buffer with fused ReLU on load.
// ---------------------------------------------------------------------------
__global__ __launch_bounds__(256) void wn_gemm(
    const __bf16* __restrict__ Apk, const float* __restrict__ bias,
    const __bf16* __restrict__ Bbf, const float* __restrict__ Bf,
    float* __restrict__ out, int Ktot)
{
    __shared__ __attribute__((aligned(16))) __bf16 As[2][64][LDA];
    __shared__ __attribute__((aligned(16))) __bf16 Bt[2][64][LDA];

    const int tid  = threadIdx.x;
    const int lane = tid & 31;
    const int wave = tid >> 5;
    const int row16   = (wave & 3) * 16;
    const int colHalf = (wave >> 2) * 32;
    const int u0 = blockIdx.x * 64;
    const int m0 = blockIdx.y * 64;

    const int a_ml  = tid >> 2;
    const int a_kl0 = (tid & 3) * 8;
    const int b_kl  = tid >> 3;
    const int b_n0  = (tid & 7) * 8;

    const __bf16* app = Apk + (size_t)(m0 + a_ml) * Ktot + a_kl0;
    const unsigned ldsA0 = lds_off(&As[0][a_ml][a_kl0]);
    const unsigned ldsA1 = lds_off(&As[1][a_ml][a_kl0]);

    v8f acc0 = {}, acc1 = {};

    auto fetch_B = [&](int p, Q8U& tb) {
        int kk = p * 32 + b_kl;
        if (Bbf) {
            tb.q = *(const uint4*)(Bbf + (size_t)kk * L_LEN + u0 + b_n0);
        } else {
            const float* src = Bf + (size_t)kk * L_LEN + u0 + b_n0;
            float4 f0 = *(const float4*)(src);
            float4 f1 = *(const float4*)(src + 4);
            tb.h[0] = (__bf16)fmaxf(f0.x, 0.f); tb.h[1] = (__bf16)fmaxf(f0.y, 0.f);
            tb.h[2] = (__bf16)fmaxf(f0.z, 0.f); tb.h[3] = (__bf16)fmaxf(f0.w, 0.f);
            tb.h[4] = (__bf16)fmaxf(f1.x, 0.f); tb.h[5] = (__bf16)fmaxf(f1.y, 0.f);
            tb.h[6] = (__bf16)fmaxf(f1.z, 0.f); tb.h[7] = (__bf16)fmaxf(f1.w, 0.f);
        }
    };
    auto store_B = [&](int buf, const Q8U& tb) {
        #pragma unroll
        for (int j = 0; j < 8; ++j) Bt[buf][b_n0 + j][b_kl] = tb.h[j];
    };
    auto compute = [&](int buf) {
        const int hh = lane >> 4;
        v16bf a  = frag_ld(&As[buf][row16 + (lane & 15)][0], hh);
        v16bf b0 = frag_ld(&Bt[buf][colHalf + (lane & 15)][0], hh);
        v16bf b1 = frag_ld(&Bt[buf][colHalf + 16 + (lane & 15)][0], hh);
        acc0 = __builtin_amdgcn_wmma_f32_16x16x32_bf16(false, a, false, b0, (short)0, acc0, false, false);
        acc1 = __builtin_amdgcn_wmma_f32_16x16x32_bf16(false, a, false, b1, (short)0, acc1, false, false);
    };

    const int NC = Ktot / 32;
    {
        async_ld_b128(ldsA0, app);
        Q8U tb; fetch_B(0, tb);
        store_B(0, tb);
        wait_async0();
    }
    __syncthreads();
    for (int p = 0; p < NC - 1; ++p) {
        async_ld_b128((p + 1) & 1 ? ldsA1 : ldsA0, app + (p + 1) * 32);
        Q8U tb; fetch_B(p + 1, tb);
        compute(p & 1);
        store_B((p + 1) & 1, tb);
        wait_async0();
        __syncthreads();
    }
    compute((NC - 1) & 1);

    const int hh = lane >> 4;
    const int nc = lane & 15;
    #pragma unroll
    for (int r = 0; r < 8; ++r) {
        int m = m0 + row16 + 8 * hh + r;
        int u = u0 + colHalf + nc;
        float b = bias[m];
        out[(size_t)m * L_LEN + u]      = acc0[r] + b;
        out[(size_t)m * L_LEN + u + 16] = acc1[r] + b;
    }
}

// ---------------------------------------------------------------------------
extern "C" void kernel_launch(void* const* d_in, const int* in_sizes, int n_in,
                              void* d_out, int out_size, void* d_ws, size_t ws_size,
                              hipStream_t stream)
{
    const float* x            = (const float*)d_in[0];
    const float* conditionals = (const float*)d_in[1];
    const float* init_w       = (const float*)d_in[2];
    const float* init_b       = (const float*)d_in[3];
    const float* init_skip_w  = (const float*)d_in[4];
    const float* init_skip_b  = (const float*)d_in[5];
    const float* filter_w     = (const float*)d_in[6];
    const float* gate_w       = (const float*)d_in[7];
    const float* skip_w       = (const float*)d_in[8];
    const float* feat_w       = (const float*)d_in[9];
    const float* fcond_w      = (const float*)d_in[10];
    const float* gcond_w      = (const float*)d_in[11];
    const float* final_skip_w = (const float*)d_in[12];
    const float* final_skip_b = (const float*)d_in[13];
    const float* final_w      = (const float*)d_in[14];
    const float* final_b      = (const float*)d_in[15];
    float* out = (float*)d_out;

    // ---- workspace carve (~43 MB total) ----
    char* p = (char*)d_ws;
    float*  feat_f32 = (float*)p;  p += (size_t)RW_C * L_LEN * 4;            // 4 MB
    float*  skip_f32 = (float*)p;  p += (size_t)SW_C * L_LEN * 4;            // 2 MB
    float*  skip2_f32= (float*)p;  p += (size_t)SW_C * L_LEN * 4;            // 2 MB
    __bf16* feat_bf  = (__bf16*)p; p += (size_t)RW_C * L_LEN * 2;            // 2 MB
    __bf16* r_bf     = (__bf16*)p; p += (size_t)RW_C * L_LEN * 2;            // 2 MB
    __bf16* cond_bf  = (__bf16*)p; p += (size_t)CDIM * L_LEN * 2;            // 128 KB
    __bf16* Wf_bf    = (__bf16*)p; p += (size_t)NL_C * RW_C * KTOT_A * 2;    // 12.3 MB
    __bf16* Wg_bf    = (__bf16*)p; p += (size_t)NL_C * RW_C * KTOT_A * 2;    // 12.3 MB
    __bf16* Wp_bf    = (__bf16*)p; p += (size_t)NL_C * (RW_C+SW_C) * RW_C*2; // 5.9 MB
    __bf16* isw_bf   = (__bf16*)p; p += (size_t)SW_C * RW_C * 2;
    __bf16* fsw_bf   = (__bf16*)p; p += (size_t)SW_C * SW_C * 2;
    __bf16* fin_bf   = (__bf16*)p; p += (size_t)RW_C * SW_C * 2;

    // ---- weight packing (runs every launch; deterministic) ----
    wn_pack_gau<<<(NL_C * RW_C * KTOT_A) / 256, 256, 0, stream>>>(
        filter_w, gate_w, fcond_w, gcond_w, Wf_bf, Wg_bf);
    wn_pack_post<<<(NL_C * (RW_C + SW_C) * RW_C) / 256, 256, 0, stream>>>(
        feat_w, skip_w, Wp_bf);
    wn_cvt<<<(SW_C * RW_C) / 256, 256, 0, stream>>>(init_skip_w, isw_bf, SW_C * RW_C);
    wn_cvt<<<(SW_C * SW_C) / 256, 256, 0, stream>>>(final_skip_w, fsw_bf, SW_C * SW_C);
    wn_cvt<<<(RW_C * SW_C) / 256, 256, 0, stream>>>(final_w, fin_bf, RW_C * SW_C);
    wn_cvt<<<(CDIM * L_LEN) / 256, 256, 0, stream>>>(conditionals, cond_bf, CDIM * L_LEN);

    // ---- init conv ----
    wn_init<<<(RW_C * L_LEN) / 256, 256, 0, stream>>>(x, init_w, init_b, feat_f32, feat_bf);

    // ---- skip = init_skip_w @ feat + b ----
    wn_gemm<<<dim3(L_LEN / 64, SW_C / 64), 256, 0, stream>>>(
        isw_bf, init_skip_b, feat_bf, nullptr, skip_f32, RW_C);

    // ---- 30 gated residual layers (serial; stream order = global barrier) ----
    for (int i = 0; i < NL_C; ++i) {
        int dil = 1 << (i % 10);
        wn_gau<<<dim3(L_LEN / 64, RW_C / 64), 256, 0, stream>>>(
            Wf_bf, Wg_bf, feat_bf, cond_bf, r_bf, i, dil);
        wn_post<<<dim3(L_LEN / 64, (RW_C + SW_C) / 64), 256, 0, stream>>>(
            Wp_bf, r_bf, feat_f32, feat_bf, skip_f32, i);
    }

    // ---- head ----
    wn_gemm<<<dim3(L_LEN / 64, SW_C / 64), 256, 0, stream>>>(
        fsw_bf, final_skip_b, nullptr, skip_f32, skip2_f32, SW_C);
    wn_gemm<<<dim3(L_LEN / 64, RW_C / 64), 256, 0, stream>>>(
        fin_bf, final_b, nullptr, skip2_f32, out, SW_C);
}
